// SPiT_37915971289114
// MI455X (gfx1250) — compile-verified, matrix operand order checked by reading, not run on hardware
//
#include <hip/hip_runtime.h>
#include <cstdint>
#include <cstddef>

typedef __attribute__((ext_vector_type(16))) __bf16 v16bf;
typedef __attribute__((ext_vector_type(8)))  __bf16 v8bf;
typedef __attribute__((ext_vector_type(8)))  float  v8f;

#define NTOK 8192
#define TT   2048
#define BB   4
#define CC   768
#define HH   12
#define DD   64
#define HIDN 3072
#define C3   2304

static __device__ __forceinline__ v16bf cat8(v8bf a, v8bf b) {
  return __builtin_shufflevector(a, b, 0,1,2,3,4,5,6,7,8,9,10,11,12,13,14,15);
}
static __device__ __forceinline__ v8f wmma_bf16(v16bf a, v16bf b, v8f c) {
  // D = A(16x32 bf16) * B(32x16 bf16) + C(16x16 f32)
  return __builtin_amdgcn_wmma_f32_16x16x32_bf16(false, a, false, b, (short)0, c, false, false);
}

// ---------------- fp32 -> bf16 weight conversion ----------------
__global__ __launch_bounds__(256) void cvt_bf16(const float* __restrict__ s,
                                                __bf16* __restrict__ d, int n) {
  int i = blockIdx.x * 256 + threadIdx.x;
  if (i < n) d[i] = (__bf16)s[i];
}

// ---------------- LayerNorm (one row of 768 per block) ----------------
__global__ __launch_bounds__(256) void ln_kernel(const float* __restrict__ x,
                                                 const float* __restrict__ g,
                                                 const float* __restrict__ b,
                                                 __bf16* __restrict__ o) {
  __shared__ float red[256];
  const int row = blockIdx.x;
  const int tid = threadIdx.x;
  const float* xr = x + (size_t)row * CC;
  float v0 = xr[tid], v1 = xr[tid + 256], v2 = xr[tid + 512];
  red[tid] = v0 + v1 + v2;
  __syncthreads();
  for (int st = 128; st > 0; st >>= 1) { if (tid < st) red[tid] += red[tid + st]; __syncthreads(); }
  float mu = red[0] * (1.0f / 768.0f);
  __syncthreads();
  float d0 = v0 - mu, d1 = v1 - mu, d2 = v2 - mu;
  red[tid] = d0 * d0 + d1 * d1 + d2 * d2;
  __syncthreads();
  for (int st = 128; st > 0; st >>= 1) { if (tid < st) red[tid] += red[tid + st]; __syncthreads(); }
  float rs = rsqrtf(red[0] * (1.0f / 768.0f) + 1e-6f);
  __bf16* orow = o + (size_t)row * CC;
  orow[tid]       = (__bf16)(d0 * rs * g[tid]       + b[tid]);
  orow[tid + 256] = (__bf16)(d1 * rs * g[tid + 256] + b[tid + 256]);
  orow[tid + 512] = (__bf16)(d2 * rs * g[tid + 512] + b[tid + 512]);
}

// ---------------- WMMA GEMM: C[M,N] = A[M,K](bf16) x B[K,N](bf16) ----------------
// Block: 256 thr (8 waves), tile 128(M) x 64(N), BK=32. Waves 4x2, each 32x32.
// MODE 0: qkv epilogue (mask, scatter into Q[B,H,T,D], K[B,H,T,D], Vt[B,H,D,T])
// MODE 1: out_f32 = resid + ls * (mask ? v + bias : 0)   (N must be 768)
// MODE 2: out_bf16 = mask ? gelu(v + bias) : 0           (N = 3072)
template <int MODE>
__global__ __launch_bounds__(256) void gemm_wmma(
    const __bf16* __restrict__ A, const __bf16* __restrict__ Bw,
    int K, int N,
    const uint8_t* __restrict__ mask,
    const float* __restrict__ bias, const float* __restrict__ ls,
    const float* __restrict__ resid, float* __restrict__ outf,
    __bf16* __restrict__ outb,
    __bf16* __restrict__ qb, __bf16* __restrict__ kbuf, __bf16* __restrict__ vtb) {
  constexpr int AS = 40, BS = 40;           // bf16 strides (80B: 16B-aligned, conflict-free)
  __shared__ __bf16 Asm[128 * AS];
  __shared__ __bf16 Bsm[64 * BS];
  const int tid = threadIdx.x;
  const int m0 = blockIdx.y * 128;
  const int n0 = blockIdx.x * 64;
  const int w = tid >> 5, lane = tid & 31;
  const int wr = w & 3, wc = w >> 2;
  const int lo = lane & 15, hi = lane >> 4, kb = hi * 8;

  v8f acc[2][2];
#pragma unroll
  for (int i = 0; i < 2; i++)
#pragma unroll
    for (int j = 0; j < 2; j++)
#pragma unroll
      for (int r = 0; r < 8; r++) acc[i][j][r] = 0.0f;

  for (int k0 = 0; k0 < K; k0 += 32) {
    // stage A tile 128x32 (row-major)
#pragma unroll
    for (int i = 0; i < 2; i++) {
      int idx = tid + i * 256;
      int row = idx >> 2, ch = idx & 3;
      *(v8bf*)(Asm + row * AS + ch * 8) =
          *(const v8bf*)(A + (size_t)(m0 + row) * K + k0 + ch * 8);
    }
    // stage B tile 32x64 transposed -> Bsm[n][k]
    {
      int kr = tid >> 3, ch = tid & 7;
      v8bf t = *(const v8bf*)(Bw + (size_t)(k0 + kr) * N + n0 + ch * 8);
#pragma unroll
      for (int e = 0; e < 8; e++) Bsm[(ch * 8 + e) * BS + kr] = t[e];
    }
    __syncthreads();

    v16bf af[2], bfr[2];
#pragma unroll
    for (int tm = 0; tm < 2; tm++) {
      int row = wr * 32 + tm * 16 + lo;
      af[tm] = cat8(*(const v8bf*)(Asm + row * AS + kb),
                    *(const v8bf*)(Asm + row * AS + kb + 16));
    }
#pragma unroll
    for (int tn = 0; tn < 2; tn++) {
      int col = wc * 32 + tn * 16 + lo;
      bfr[tn] = cat8(*(const v8bf*)(Bsm + col * BS + kb),
                     *(const v8bf*)(Bsm + col * BS + kb + 16));
    }
#pragma unroll
    for (int tm = 0; tm < 2; tm++)
#pragma unroll
      for (int tn = 0; tn < 2; tn++)
        acc[tm][tn] = wmma_bf16(af[tm], bfr[tn], acc[tm][tn]);
    __syncthreads();
  }

  // epilogue
#pragma unroll
  for (int tm = 0; tm < 2; tm++) {
#pragma unroll
    for (int tn = 0; tn < 2; tn++) {
      int colg = n0 + wc * 32 + tn * 16 + lo;
#pragma unroll
      for (int r = 0; r < 8; r++) {
        int rg = m0 + wr * 32 + tm * 16 + hi * 8 + r;   // token index
        float v = acc[tm][tn][r];
        float mk = mask[rg] ? 1.0f : 0.0f;
        if (MODE == 0) {
          int s = colg / 768, rem = colg - s * 768;
          int h = rem >> 6, d = rem & 63;
          int bi = rg >> 11, t = rg & 2047;              // T = 2048
          __bf16 bv = (__bf16)(v * mk);
          size_t bh = (size_t)bi * HH + h;
          if (s == 0)      qb  [(bh * TT + t) * DD + d] = bv;
          else if (s == 1) kbuf[(bh * TT + t) * DD + d] = bv;
          else             vtb [(bh * DD + d) * TT + t] = bv;   // V transposed
        } else if (MODE == 1) {
          float yv = mk * (v + bias[colg]);
          outf[(size_t)rg * CC + colg] = resid[(size_t)rg * CC + colg] + ls[colg] * yv;
        } else {
          float tv = v + bias[colg];
          float gl = 0.5f * tv * (1.0f + erff(tv * 0.70710678118f));
          outb[(size_t)rg * HIDN + colg] = (__bf16)(mk * gl);
        }
      }
    }
  }
}

// ---------------- Flash attention: one (b,h), 128 q-rows per block ----------------
__global__ __launch_bounds__(256) void attn_wmma(
    const __bf16* __restrict__ Q, const __bf16* __restrict__ Km,
    const __bf16* __restrict__ Vt, __bf16* __restrict__ O) {
  __shared__ __bf16 P[8 * 16 * 40];   // per-wave 16x32 P tile, stride 40
  const int bh = blockIdx.x;
  const int q0 = blockIdx.y * 128;
  const int w = threadIdx.x >> 5, lane = threadIdx.x & 31;
  const int lo = lane & 15, hi = lane >> 4, kb = hi * 8;
  __bf16* Pw = P + w * 16 * 40;
  const int qw = q0 + w * 16;
  const __bf16* Qp = Q + ((size_t)bh * TT + qw) * DD;
  const __bf16* Kp = Km + (size_t)bh * TT * DD;
  const __bf16* Vp = Vt + (size_t)bh * DD * TT;

  v16bf qf[2];
#pragma unroll
  for (int c = 0; c < 2; c++)
    qf[c] = cat8(*(const v8bf*)(Qp + lo * DD + c * 32 + kb),
                 *(const v8bf*)(Qp + lo * DD + c * 32 + kb + 16));

  float m[8], l[8];
  v8f oacc[4];
#pragma unroll
  for (int r = 0; r < 8; r++) { m[r] = -1e30f; l[r] = 0.0f; }
#pragma unroll
  for (int t = 0; t < 4; t++)
#pragma unroll
    for (int r = 0; r < 8; r++) oacc[t][r] = 0.0f;

  for (int key0 = 0; key0 < TT; key0 += 32) {
    v8f sa[2];
#pragma unroll
    for (int st = 0; st < 2; st++) {
#pragma unroll
      for (int r = 0; r < 8; r++) sa[st][r] = 0.0f;
      const __bf16* kr = Kp + (size_t)(key0 + st * 16 + lo) * DD;
#pragma unroll
      for (int c = 0; c < 2; c++) {
        v16bf kf = cat8(*(const v8bf*)(kr + c * 32 + kb),
                        *(const v8bf*)(kr + c * 32 + kb + 16));
        sa[st] = wmma_bf16(qf[c], kf, sa[st]);
      }
    }
    float rsf[8];
#pragma unroll
    for (int r = 0; r < 8; r++) {
      float s0 = sa[0][r] * 0.125f;          // D^-0.5
      float s1 = sa[1][r] * 0.125f;
      float x = fmaxf(s0, s1);
      x = fmaxf(x, __shfl_xor(x, 1, 32));
      x = fmaxf(x, __shfl_xor(x, 2, 32));
      x = fmaxf(x, __shfl_xor(x, 4, 32));
      x = fmaxf(x, __shfl_xor(x, 8, 32));
      float mn = fmaxf(m[r], x);
      float p0 = __expf(s0 - mn);
      float p1 = __expf(s1 - mn);
      sa[0][r] = p0; sa[1][r] = p1;
      float ssum = p0 + p1;
      ssum += __shfl_xor(ssum, 1, 32);
      ssum += __shfl_xor(ssum, 2, 32);
      ssum += __shfl_xor(ssum, 4, 32);
      ssum += __shfl_xor(ssum, 8, 32);
      float corr = __expf(m[r] - mn);
      l[r] = l[r] * corr + ssum;
      m[r] = mn;
      rsf[r] = corr;
    }
#pragma unroll
    for (int t = 0; t < 4; t++)
#pragma unroll
      for (int r = 0; r < 8; r++) oacc[t][r] *= rsf[r];
    // C/D layout -> A layout via per-wave LDS tile (LDS ops are in-order per wave)
#pragma unroll
    for (int st = 0; st < 2; st++)
#pragma unroll
      for (int r = 0; r < 8; r++)
        Pw[(r + hi * 8) * 40 + st * 16 + lo] = (__bf16)sa[st][r];
    v16bf pf = cat8(*(const v8bf*)(Pw + lo * 40 + kb),
                    *(const v8bf*)(Pw + lo * 40 + kb + 16));
#pragma unroll
    for (int t = 0; t < 4; t++) {
      const __bf16* vr = Vp + (size_t)(t * 16 + lo) * TT + key0;  // Vt row = d, contiguous in keys
      v16bf vf = cat8(*(const v8bf*)(vr + kb), *(const v8bf*)(vr + kb + 16));
      oacc[t] = wmma_bf16(pf, vf, oacc[t]);
    }
  }
  const int bi = bh / HH, h = bh - bi * HH;
#pragma unroll
  for (int t = 0; t < 4; t++)
#pragma unroll
    for (int r = 0; r < 8; r++) {
      int row = qw + r + hi * 8;
      float val = oacc[t][r] / l[r];
      O[((size_t)bi * TT + row) * CC + h * DD + t * 16 + lo] = (__bf16)val;
    }
}

extern "C" void kernel_launch(void* const* d_in, const int* in_sizes, int n_in,
                              void* d_out, int out_size, void* d_ws, size_t ws_size,
                              hipStream_t stream) {
  (void)in_sizes; (void)n_in; (void)out_size; (void)ws_size;
  const float*   x      = (const float*)d_in[0];
  const uint8_t* am     = (const uint8_t*)d_in[1];
  const float*   n1g    = (const float*)d_in[2];
  const float*   n1b    = (const float*)d_in[3];
  const float*   n2g    = (const float*)d_in[4];
  const float*   n2b    = (const float*)d_in[5];
  const float*   ls1    = (const float*)d_in[6];
  const float*   ls2    = (const float*)d_in[7];
  const float*   w_qkv  = (const float*)d_in[8];
  const float*   w_proj = (const float*)d_in[9];
  const float*   b_proj = (const float*)d_in[10];
  const float*   w1     = (const float*)d_in[11];
  const float*   b1     = (const float*)d_in[12];
  const float*   w2     = (const float*)d_in[13];
  const float*   b2     = (const float*)d_in[14];
  float* out = (float*)d_out;

  char* p = (char*)d_ws;
  auto carve = [&](size_t n) -> char* {
    char* r = p; p += (n + 255) & ~(size_t)255; return r;
  };
  __bf16* wqkv_bf  = (__bf16*)carve((size_t)CC * C3 * 2);
  __bf16* wproj_bf = (__bf16*)carve((size_t)CC * CC * 2);
  __bf16* w1_bf    = (__bf16*)carve((size_t)CC * HIDN * 2);
  __bf16* w2_bf    = (__bf16*)carve((size_t)HIDN * CC * 2);
  __bf16* h_bf     = (__bf16*)carve((size_t)NTOK * CC * 2);   // reused for LN2 output
  float*  x1       = (float*) carve((size_t)NTOK * CC * 4);
  char*   big      = carve((size_t)NTOK * HIDN * 2);          // qkv/attn region, reused as MLP hidden
  __bf16* qb  = (__bf16*)big;
  __bf16* kb  = (__bf16*)(big + 1 * (size_t)NTOK * CC * 2);
  __bf16* vtb = (__bf16*)(big + 2 * (size_t)NTOK * CC * 2);
  __bf16* ob  = (__bf16*)(big + 3 * (size_t)NTOK * CC * 2);
  __bf16* yb  = (__bf16*)big;

  // weights -> bf16
  cvt_bf16<<<(CC * C3 + 255) / 256, 256, 0, stream>>>(w_qkv, wqkv_bf, CC * C3);
  cvt_bf16<<<(CC * CC + 255) / 256, 256, 0, stream>>>(w_proj, wproj_bf, CC * CC);
  cvt_bf16<<<(CC * HIDN + 255) / 256, 256, 0, stream>>>(w1, w1_bf, CC * HIDN);
  cvt_bf16<<<(HIDN * CC + 255) / 256, 256, 0, stream>>>(w2, w2_bf, HIDN * CC);

  // LN1 -> h
  ln_kernel<<<NTOK, 256, 0, stream>>>(x, n1g, n1b, h_bf);
  // QKV GEMM (masked), scatter into Q/K/Vt head layouts
  gemm_wmma<0><<<dim3(C3 / 64, NTOK / 128), 256, 0, stream>>>(
      h_bf, wqkv_bf, CC, C3, am, nullptr, nullptr, nullptr, nullptr, nullptr, qb, kb, vtb);
  // attention
  attn_wmma<<<dim3(BB * HH, TT / 128), 256, 0, stream>>>(qb, kb, vtb, ob);
  // proj + residual -> x1
  gemm_wmma<1><<<dim3(CC / 64, NTOK / 128), 256, 0, stream>>>(
      ob, wproj_bf, CC, CC, am, b_proj, ls1, x, x1, nullptr, nullptr, nullptr, nullptr);
  // LN2 -> h (reuse)
  ln_kernel<<<NTOK, 256, 0, stream>>>(x1, n2g, n2b, h_bf);
  // MLP1 (+gelu, mask) -> yb
  gemm_wmma<2><<<dim3(HIDN / 64, NTOK / 128), 256, 0, stream>>>(
      h_bf, w1_bf, CC, HIDN, am, b1, nullptr, nullptr, nullptr, yb, nullptr, nullptr, nullptr);
  // MLP2 + residual -> out
  gemm_wmma<1><<<dim3(CC / 64, NTOK / 128), 256, 0, stream>>>(
      yb, w2_bf, HIDN, CC, am, b2, ls2, x1, out, nullptr, nullptr, nullptr, nullptr);
}
